// QuantLinearFp8Handler_2499670966504
// MI455X (gfx1250) — 8-bit matrix operand order repaired and checked by reading, compiled, not run on hardware
//
#include <hip/hip_runtime.h>
#include <hip/hip_bf16.h>


typedef __attribute__((ext_vector_type(16))) int   v16i;
typedef __attribute__((ext_vector_type(8)))  float v8f;
typedef __attribute__((ext_vector_type(4)))  unsigned int u32x4;
typedef __attribute__((ext_vector_type(8)))  int   i32x8;
typedef __attribute__((ext_vector_type(4)))  int   i32x4;

#define FP8_MAX 448.0f

#if defined(__has_builtin)
#if __has_builtin(__builtin_amdgcn_tensor_load_to_lds) && \
    __has_builtin(__builtin_amdgcn_s_wait_tensorcnt)
#define USE_TDM 1
#endif
#endif


__device__ __forceinline__ unsigned int f32_to_e4m3_byte(float f) {
  unsigned int u    = __float_as_uint(f);
  unsigned int sign = (u >> 31) << 7;
  unsigned int absu = u & 0x7fffffffu;
  if (absu < 0x3C800000u) {
    float af = __uint_as_float(absu);
    unsigned int m = (unsigned int)__builtin_rintf(af * 512.0f);
    return sign | m;
  }
  unsigned int lsb = (absu >> 20) & 1u;
  absu += 0x7FFFFu + lsb;
  unsigned int e = (absu >> 23) - 120u;
  unsigned int m = (absu >> 20) & 0x7u;
  return sign | (e << 3) | m;
}

__global__ __launch_bounds__(256)
void quant_fp8_kernel(const float* __restrict__ in,
                      unsigned char* __restrict__ out,
                      const float* __restrict__ scale,
                      long long n4) {
  long long i = (long long)blockIdx.x * blockDim.x + threadIdx.x;
  if (i >= n4) return;
  const float inv = 1.0f / scale[0];
  float4 v = ((const float4*)in)[i];
  float a = fminf(fmaxf(v.x * inv, -FP8_MAX), FP8_MAX);
  float b = fminf(fmaxf(v.y * inv, -FP8_MAX), FP8_MAX);
  float c = fminf(fmaxf(v.z * inv, -FP8_MAX), FP8_MAX);
  float d = fminf(fmaxf(v.w * inv, -FP8_MAX), FP8_MAX);
#if __has_builtin(__builtin_amdgcn_cvt_pk_fp8_f32)
  int packed = 0;
  packed = __builtin_amdgcn_cvt_pk_fp8_f32(a, b, packed, false);
  packed = __builtin_amdgcn_cvt_pk_fp8_f32(c, d, packed, true);
  unsigned int p = (unsigned int)packed;
#else
  unsigned int p = f32_to_e4m3_byte(a)
                 | (f32_to_e4m3_byte(b) << 8)
                 | (f32_to_e4m3_byte(c) << 16)
                 | (f32_to_e4m3_byte(d) << 24);
#endif
  ((unsigned int*)out)[i] = p;
}


#define BM 128
#define BN 128
#define BK 128
#define LDA 144

__global__ __launch_bounds__(256)
void fp8_gemm_wmma_kernel(const unsigned char* __restrict__ Aq,
                          const unsigned char* __restrict__ Bq,
                          const float* __restrict__ bias,
                          const float* __restrict__ scale_in,
                          const float* __restrict__ scale_w,
                          _Float16* __restrict__ out,
                          int M, int N, int K) {
  __shared__ unsigned char ldsA[2][BM * LDA];

  const int tid  = threadIdx.x;
  const int lane = tid & 31;
  const int wave = tid >> 5;
  const int wm   = wave & 3;
  const int wn   = wave >> 2;
  const int half = lane >> 4;
  const int l16  = lane & 15;

  const int nBase = blockIdx.x * BN;
  const int mBase = blockIdx.y * BM;
  const unsigned char* Abase = Aq + (size_t)mBase * K;

#ifdef USE_TDM
  const unsigned ldsOff[2] = { (unsigned)(size_t)&ldsA[0][0],
                               (unsigned)(size_t)&ldsA[1][0] };
  auto tdm_issue_A = [&](int buf, int k0) {
    unsigned long long ga = (unsigned long long)(Abase + k0);
    u32x4 g0;
    g0[0] = 1u;
    g0[1] = ldsOff[buf];
    g0[2] = (unsigned)(ga & 0xffffffffu);
    g0[3] = ((unsigned)(ga >> 32) & 0x01ffffffu)
            | (2u << 30);
    i32x8 g1;
    g1[0] = (int)((3u << 16)
                | (1u << 20)
                | (4u << 22)
                | (3u << 25));
    g1[1] = (int)(256u << 16);
    g1[2] = (int)(128u << 16);
    g1[3] = (int)(16u  << 16);
    g1[4] = 128;
    g1[5] = 256;
    g1[6] = 0;
    g1[7] = 0;
    i32x4 z4 = { 0, 0, 0, 0 };
    i32x8 z8 = { 0, 0, 0, 0, 0, 0, 0, 0 };
    __builtin_amdgcn_tensor_load_to_lds(g0, g1, z4, z4, z8, 0);
  };
#else
  auto stage_A = [&](int buf, int k0) {
#pragma unroll
    for (int p = 0; p < 4; ++p) {
      int idx  = p * 256 + tid;
      int row  = idx >> 3;
      int colb = (idx & 7) << 4;
      uint4 v = *(const uint4*)(Abase + (size_t)row * K + k0 + colb);
      *(uint4*)(&ldsA[buf][0] + row * LDA + colb) = v;
    }
  };
#endif

  v8f acc[2][4];
#pragma unroll
  for (int mi = 0; mi < 2; ++mi)
#pragma unroll
    for (int ni = 0; ni < 4; ++ni)
#pragma unroll
      for (int r = 0; r < 8; ++r) acc[mi][ni][r] = 0.0f;

#ifdef USE_TDM
  if (wave == 0) {
    tdm_issue_A(0, 0);
    __builtin_amdgcn_s_wait_tensorcnt(0);
  }
#else
  stage_A(0, 0);
#endif
  __syncthreads();

  const int nk = K / BK;
  for (int i = 0; i < nk; ++i) {
    const int k0  = i * BK;
    const int cur = i & 1;

    if (i + 1 < nk) {
#ifdef USE_TDM
      if (wave == 0) tdm_issue_A(cur ^ 1, k0 + BK);
#else
      stage_A(cur ^ 1, k0 + BK);
#endif
    }

    v16i bfrag[4];
#pragma unroll
    for (int ni = 0; ni < 4; ++ni) {
      const int col = nBase + wn * 64 + ni * 16 + l16;
      const unsigned char* bp = Bq + (size_t)col * K + k0 + half * 16;
#pragma unroll
      for (int q = 0; q < 4; ++q) {
        uint4 d = *(const uint4*)(bp + q * 32);
        bfrag[ni][4 * q + 0] = (int)d.x;
        bfrag[ni][4 * q + 1] = (int)d.y;
        bfrag[ni][4 * q + 2] = (int)d.z;
        bfrag[ni][4 * q + 3] = (int)d.w;
      }
      if (k0 + BK < K) __builtin_prefetch(bp + BK, 0, 3);
    }

    const unsigned char* As = &ldsA[cur][0];
#pragma unroll
    for (int mi = 0; mi < 2; ++mi) {
      const int row = wm * 32 + mi * 16 + l16;
      v16i afrag;
#pragma unroll
      for (int c = 0; c < 8; ++c) {
        int koff = ((c >> 1) << 5) + half * 16 + ((c & 1) << 3);
        uint2 d = *(const uint2*)(As + row * LDA + koff);
        afrag[2 * c + 0] = (int)d.x;
        afrag[2 * c + 1] = (int)d.y;
      }
#pragma unroll
      for (int ni = 0; ni < 4; ++ni) {
        acc[mi][ni] = __builtin_amdgcn_wmma_f32_16x16x128_fp8_fp8(
            afrag, bfrag[ni], (short)0, acc[mi][ni],
             false,  false);
      }
    }

    if (i + 1 < nk) {
#ifdef USE_TDM
      if (wave == 0) __builtin_amdgcn_s_wait_tensorcnt(0);
#endif
      __syncthreads();
    }
  }

  const float s = scale_in[0] * scale_w[0];
#pragma unroll
  for (int mi = 0; mi < 2; ++mi) {
#pragma unroll
    for (int ni = 0; ni < 4; ++ni) {
      const int gcol = nBase + wn * 64 + ni * 16 + l16;
      const _Float16 bv = (_Float16)bias[gcol];
#pragma unroll
      for (int r = 0; r < 8; ++r) {
        const int grow = mBase + wm * 32 + mi * 16 + r + half * 8;
        _Float16 o = (_Float16)(acc[mi][ni][r] * s) + bv;
        out[(size_t)grow * N + gcol] = o;
      }
    }
  }
}


extern "C" void kernel_launch(void* const* d_in, const int* in_sizes, int n_in,
                              void* d_out, int out_size, void* d_ws, size_t ws_size,
                              hipStream_t stream) {
  const float* x    = (const float*)d_in[0];
  const float* w    = (const float*)d_in[1];
  const float* bias = (const float*)d_in[2];
  const float* sIn  = (const float*)d_in[3];
  const float* sW   = (const float*)d_in[4];

  const int K = 2048;
  const int N = in_sizes[2];
  const long long xElems = (long long)in_sizes[0];
  const long long wElems = (long long)in_sizes[1];
  const int M = (int)(xElems / K);

  unsigned char* xq = (unsigned char*)d_ws;
  unsigned char* wq = xq + (size_t)M * K;

  const long long nx4 = xElems / 4;
  const long long nw4 = wElems / 4;
  quant_fp8_kernel<<<(unsigned)((nx4 + 255) / 256), 256, 0, stream>>>(x, xq, sIn, nx4);
  quant_fp8_kernel<<<(unsigned)((nw4 + 255) / 256), 256, 0, stream>>>(w, wq, sW, nw4);

  dim3 grid(N / BN, M / BM);
  fp8_gemm_wmma_kernel<<<grid, 256, 0, stream>>>(
      xq, wq, bias, sIn, sW, (_Float16*)d_out, M, N, K);
}
